// MultiHeadAttention_64132451663888
// MI455X (gfx1250) — matrix operand order repaired and checked by reading, not compiled, not run on hardware
//
#include <hip/hip_runtime.h>
#include <hip/hip_bf16.h>

// ---------------------------------------------------------------------------
// MHA forward for gfx1250 (CDNA5, wave32, WMMA 16x16x32 f16, f32 accumulate)
// d_out = [ out (B*S*1024 f32) | attn (B*H*S*S f32) ]
// ---------------------------------------------------------------------------

#define DIM 1024
#define NHEAD 16
#define DHEAD 64
#define BB 2
#define SS 2048
#define MROWS (BB * SS) // 4096

typedef __attribute__((ext_vector_type(16))) _Float16 v16h;
typedef __attribute__((ext_vector_type(8)))  _Float16 v8h;
typedef __attribute__((ext_vector_type(4)))  _Float16 v4h;
typedef __attribute__((ext_vector_type(8)))  float    v8f;

static __device__ inline v8h ld8(const _Float16* p) { return *(const v8h*)p; }

static __device__ inline v16h cat16(v8h lo, v8h hi) {
  v16h r;
#pragma unroll
  for (int i = 0; i < 8; ++i) { r[i] = lo[i]; r[i + 8] = hi[i]; }
  return r;
}

static __device__ inline v8f wmma_f16(v16h a, v16h b, v8f c) {
  return __builtin_amdgcn_wmma_f32_16x16x32_f16(false, a, false, b, (short)0, c,
                                                false, false);
}

// ---------------------------------------------------------------------------
// Kernel 1: QKV projections.  P = X @ W^T + bias  (fp32 in, f16 out)
// grid = (M/64, N/64, 3); block = 256 (8 waves), each wave -> 16x32 out tile.
// z==2 (V) stores transposed as Vt[b*1024 + n][s]  (== [B,H,D,S]).
// ---------------------------------------------------------------------------
__global__ __launch_bounds__(256) void mha_qkv_kernel(
    const float* __restrict__ xq, const float* __restrict__ xk,
    const float* __restrict__ xv, const float* __restrict__ wq,
    const float* __restrict__ wk, const float* __restrict__ wv,
    const float* __restrict__ bq, const float* __restrict__ bk,
    const float* __restrict__ bv, _Float16* __restrict__ q_ws,
    _Float16* __restrict__ k_ws, _Float16* __restrict__ vt_ws) {
  __shared__ _Float16 As[64][40];
  __shared__ _Float16 Bs[64][40];

  const int z = blockIdx.z;
  const float* X    = (z == 0) ? xq : (z == 1) ? xk : xv;
  const float* W    = (z == 0) ? wq : (z == 1) ? wk : wv;
  const float* bias = (z == 0) ? bq : (z == 1) ? bk : bv;

  const int m0 = blockIdx.x * 64;
  const int n0 = blockIdx.y * 64;
  const int t = threadIdx.x;
  const int l = t & 31;
  const int w = t >> 5;
  const int wm = w >> 1;          // 0..3
  const int wn = w & 1;           // 0..1
  const int lm = l & 15;
  const int hi = (l >= 16);

  v8f c0 = {}, c1 = {};
  const int ldrow = t >> 3;              // 0..31
  const int col4  = (t & 7) * 4;         // 0..28

  for (int k0 = 0; k0 < DIM; k0 += 32) {
    __syncthreads();
#pragma unroll
    for (int rr = 0; rr < 2; ++rr) {
      const int r = ldrow + rr * 32;
      float4 a = *(const float4*)(X + (size_t)(m0 + r) * DIM + k0 + col4);
      v4h ah = {(_Float16)a.x, (_Float16)a.y, (_Float16)a.z, (_Float16)a.w};
      *(v4h*)(&As[r][col4]) = ah;
      float4 b = *(const float4*)(W + (size_t)(n0 + r) * DIM + k0 + col4);
      v4h bh = {(_Float16)b.x, (_Float16)b.y, (_Float16)b.z, (_Float16)b.w};
      *(v4h*)(&Bs[r][col4]) = bh;
    }
    __syncthreads();

    const int arow = wm * 16 + lm;
    const int alo  = hi ? 8 : 0;
    v16h af = cat16(ld8(&As[arow][alo]), ld8(&As[arow][16 + alo]));

    const int klo   = hi ? 8 : 0;
    const int brow0 = wn * 32 + lm;
    v16h b0 = cat16(ld8(&Bs[brow0][klo]), ld8(&Bs[brow0][klo + 16]));
    v16h b1 = cat16(ld8(&Bs[brow0 + 16][klo]), ld8(&Bs[brow0 + 16][klo + 16]));

    c0 = wmma_f16(af, b0, c0);
    c1 = wmma_f16(af, b1, c1);
  }

  const int mbase = m0 + wm * 16 + (hi ? 8 : 0);
  const int ncol0 = n0 + wn * 32 + lm;
  const float bia0 = bias[ncol0];
  const float bia1 = bias[ncol0 + 16];
#pragma unroll
  for (int r = 0; r < 8; ++r) {
    const int m = mbase + r;
    const float v0 = c0[r] + bia0;
    const float v1 = c1[r] + bia1;
    if (z == 2) {
      const int bb = m >> 11, s = m & 2047;
      vt_ws[((size_t)(bb * DIM + ncol0)) * SS + s]        = (_Float16)v0;
      vt_ws[((size_t)(bb * DIM + ncol0 + 16)) * SS + s]   = (_Float16)v1;
    } else {
      _Float16* O = (z == 0) ? q_ws : k_ws;
      O[(size_t)m * DIM + ncol0]      = (_Float16)v0;
      O[(size_t)m * DIM + ncol0 + 16] = (_Float16)v1;
    }
  }
}

// ---------------------------------------------------------------------------
// Kernel 2: attention core.  grid = (S/128, B*H); block = 256 (8 waves).
// Each wave owns 16 query rows of one (b,h).
//  phase 1: scores = QK^T/8 -> attn region (raw), track row max
//  phase 2: p = exp(s-max) -> attn region (unnormalized), sum, reps += p @ V
//  phase 3: attn *= 1/sum;  reps/sum -> f16 workspace
// ---------------------------------------------------------------------------
__global__ __launch_bounds__(256) void mha_attn_kernel(
    const _Float16* __restrict__ q_ws, const _Float16* __restrict__ k_ws,
    const _Float16* __restrict__ vt_ws, _Float16* __restrict__ reps_ws,
    float* __restrict__ attn) {
  __shared__ float smax[8][16];
  __shared__ float ssum[8][16];

  const int t = threadIdx.x;
  const int l = t & 31;
  const int w = t >> 5;
  const int lm = l & 15;
  const int hi = (l >= 16);
  const int bh = blockIdx.y;
  const int b = bh >> 4;
  const int q0 = blockIdx.x * 128 + w * 16;

  const _Float16* Q = q_ws + ((size_t)(b * SS + q0)) * DIM + (bh & 15) * DHEAD;
  const _Float16* K = k_ws + ((size_t)b * SS) * DIM + (bh & 15) * DHEAD;
  const _Float16* V = vt_ws + (size_t)bh * DHEAD * SS;
  float* A = attn + ((size_t)bh * SS + q0) * SS;

  // Q fragments, reused for the entire key loop (A operand: M=q, K=d)
  const int alo = hi ? 8 : 0;
  const int klo = hi ? 8 : 0;
  const _Float16* qr = Q + (size_t)lm * DIM;
  const v16h aq0 = cat16(ld8(qr + alo), ld8(qr + 16 + alo));
  const v16h aq1 = cat16(ld8(qr + 32 + alo), ld8(qr + 48 + alo));

  // ---- phase 1: raw scores + row max ----
  float rmax[8];
#pragma unroll
  for (int r = 0; r < 8; ++r) rmax[r] = -3.0e38f;

  for (int kb = 0; kb < SS / 16; ++kb) {
    const _Float16* kr = K + (size_t)(kb * 16 + lm) * DIM;
    if (kb + 1 < SS / 16)
      __builtin_prefetch(K + (size_t)((kb + 1) * 16 + lm) * DIM, 0, 1);
    v16h bk0 = cat16(ld8(kr + klo), ld8(kr + klo + 16));
    v16h bk1 = cat16(ld8(kr + 32 + klo), ld8(kr + 32 + klo + 16));
    v8f c = {};
    c = wmma_f16(aq0, bk0, c);
    c = wmma_f16(aq1, bk1, c);
    float* Ar = A + kb * 16 + lm;
#pragma unroll
    for (int r = 0; r < 8; ++r) {
      const float v = c[r] * 0.125f;  // 1/sqrt(64)
      rmax[r] = fmaxf(rmax[r], v);
      Ar[(size_t)(r + (hi ? 8 : 0)) * SS] = v;
    }
  }
#pragma unroll
  for (int r = 0; r < 8; ++r) {
    float m = rmax[r];
    m = fmaxf(m, __shfl_xor(m, 1, 32));
    m = fmaxf(m, __shfl_xor(m, 2, 32));
    m = fmaxf(m, __shfl_xor(m, 4, 32));
    m = fmaxf(m, __shfl_xor(m, 8, 32));
    smax[w][r + (hi ? 8 : 0)] = m;
  }
  __threadfence();

  // ---- phase 2: exp, sum, reps = p @ V ----
  const float mymax = smax[w][lm];
  float rsum = 0.0f;
  v8f cr0 = {}, cr1 = {}, cr2 = {}, cr3 = {};
  for (int kb = 0; kb < SS / 32; ++kb) {
    float* Sr = A + (size_t)lm * SS + kb * 32;
    v16h ap;
#pragma unroll
    for (int j = 0; j < 8; ++j) {
      const int kbase = ((j < 4) ? 2 * j : 16 + 2 * (j - 4)) + (hi ? 8 : 0);
      float2 s2 = *(float2*)(Sr + kbase);
      const float p0 = __expf(s2.x - mymax);
      const float p1 = __expf(s2.y - mymax);
      rsum += p0 + p1;
      *(float2*)(Sr + kbase) = make_float2(p0, p1);  // unnormalized for now
      ap[2 * j] = (_Float16)p0;
      ap[2 * j + 1] = (_Float16)p1;
    }
    const size_t vofs = (size_t)kb * 32 + klo;
    const _Float16* v0 = V + (size_t)(0 * 16 + lm) * SS + vofs;
    const _Float16* v1 = V + (size_t)(1 * 16 + lm) * SS + vofs;
    const _Float16* v2 = V + (size_t)(2 * 16 + lm) * SS + vofs;
    const _Float16* v3 = V + (size_t)(3 * 16 + lm) * SS + vofs;
    cr0 = wmma_f16(ap, cat16(ld8(v0), ld8(v0 + 16)), cr0);
    cr1 = wmma_f16(ap, cat16(ld8(v1), ld8(v1 + 16)), cr1);
    cr2 = wmma_f16(ap, cat16(ld8(v2), ld8(v2 + 16)), cr2);
    cr3 = wmma_f16(ap, cat16(ld8(v3), ld8(v3 + 16)), cr3);
  }
  rsum += __shfl_xor(rsum, 16, 32);
  ssum[w][lm] = rsum;
  __threadfence();

  // ---- phase 3a: normalize attn in place ----
  for (int r = 0; r < 16; ++r) {
    const float inv = 1.0f / ssum[w][r];
    float4* Ar = (float4*)(A + (size_t)r * SS);
    for (int c4 = l; c4 < SS / 4; c4 += 32) {
      float4 v = Ar[c4];
      v.x *= inv; v.y *= inv; v.z *= inv; v.w *= inv;
      Ar[c4] = v;
    }
  }
  // ---- phase 3b: reps /= sum -> f16 workspace ----
  const int hcol = (bh & 15) * DHEAD + lm;
#pragma unroll
  for (int r = 0; r < 8; ++r) {
    const int row = r + (hi ? 8 : 0);
    const float inv = 1.0f / ssum[w][row];
    const size_t base = ((size_t)(b * SS + q0 + row)) * DIM + hcol;
    reps_ws[base + 0]  = (_Float16)(cr0[r] * inv);
    reps_ws[base + 16] = (_Float16)(cr1[r] * inv);
    reps_ws[base + 32] = (_Float16)(cr2[r] * inv);
    reps_ws[base + 48] = (_Float16)(cr3[r] * inv);
  }
}

// ---------------------------------------------------------------------------
// Kernel 3: out = reps @ Wo^T + bo   (f16 A, fp32 W converted, fp32 out)
// ---------------------------------------------------------------------------
__global__ __launch_bounds__(256) void mha_out_kernel(
    const _Float16* __restrict__ reps, const float* __restrict__ wo,
    const float* __restrict__ bo, float* __restrict__ out) {
  __shared__ _Float16 As[64][40];
  __shared__ _Float16 Bs[64][40];

  const int m0 = blockIdx.x * 64;
  const int n0 = blockIdx.y * 64;
  const int t = threadIdx.x;
  const int l = t & 31;
  const int w = t >> 5;
  const int wm = w >> 1;
  const int wn = w & 1;
  const int lm = l & 15;
  const int hi = (l >= 16);

  v8f c0 = {}, c1 = {};
  const int ldrow = t >> 3;
  const int col4  = (t & 7) * 4;

  for (int k0 = 0; k0 < DIM; k0 += 32) {
    __syncthreads();
#pragma unroll
    for (int rr = 0; rr < 2; ++rr) {
      const int r = ldrow + rr * 32;
      v4h ah = *(const v4h*)(reps + (size_t)(m0 + r) * DIM + k0 + col4);
      *(v4h*)(&As[r][col4]) = ah;
      float4 b = *(const float4*)(wo + (size_t)(n0 + r) * DIM + k0 + col4);
      v4h bh = {(_Float16)b.x, (_Float16)b.y, (_Float16)b.z, (_Float16)b.w};
      *(v4h*)(&Bs[r][col4]) = bh;
    }
    __syncthreads();

    const int arow = wm * 16 + lm;
    const int alo  = hi ? 8 : 0;
    v16h af = cat16(ld8(&As[arow][alo]), ld8(&As[arow][16 + alo]));
    const int klo   = hi ? 8 : 0;
    const int brow0 = wn * 32 + lm;
    v16h b0 = cat16(ld8(&Bs[brow0][klo]), ld8(&Bs[brow0][klo + 16]));
    v16h b1 = cat16(ld8(&Bs[brow0 + 16][klo]), ld8(&Bs[brow0 + 16][klo + 16]));
    c0 = wmma_f16(af, b0, c0);
    c1 = wmma_f16(af, b1, c1);
  }

  const int mbase = m0 + wm * 16 + (hi ? 8 : 0);
  const int ncol0 = n0 + wn * 32 + lm;
  const float bia0 = bo[ncol0];
  const float bia1 = bo[ncol0 + 16];
#pragma unroll
  for (int r = 0; r < 8; ++r) {
    const int m = mbase + r;
    out[(size_t)m * DIM + ncol0]      = c0[r] + bia0;
    out[(size_t)m * DIM + ncol0 + 16] = c1[r] + bia1;
  }
}

// ---------------------------------------------------------------------------
extern "C" void kernel_launch(void* const* d_in, const int* in_sizes, int n_in,
                              void* d_out, int out_size, void* d_ws,
                              size_t ws_size, hipStream_t stream) {
  const float* xq = (const float*)d_in[0];
  const float* xk = (const float*)d_in[1];
  const float* xv = (const float*)d_in[2];
  const float* wq = (const float*)d_in[3];
  const float* bq = (const float*)d_in[4];
  const float* wk = (const float*)d_in[5];
  const float* bk = (const float*)d_in[6];
  const float* wv = (const float*)d_in[7];
  const float* bv = (const float*)d_in[8];
  const float* wo = (const float*)d_in[9];
  const float* bo = (const float*)d_in[10];

  _Float16* q_ws    = (_Float16*)d_ws;
  _Float16* k_ws    = q_ws + (size_t)MROWS * DIM;
  _Float16* vt_ws   = k_ws + (size_t)MROWS * DIM;
  _Float16* reps_ws = vt_ws + (size_t)MROWS * DIM;

  float* out  = (float*)d_out;
  float* attn = out + (size_t)MROWS * DIM;  // attn region after `out`

  mha_qkv_kernel<<<dim3(MROWS / 64, DIM / 64, 3), 256, 0, stream>>>(
      xq, xk, xv, wq, wk, wv, bq, bk, bv, q_ws, k_ws, vt_ws);
  mha_attn_kernel<<<dim3(SS / 128, BB * NHEAD), 256, 0, stream>>>(
      q_ws, k_ws, vt_ws, reps_ws, attn);
  mha_out_kernel<<<dim3(MROWS / 64, DIM / 64), 256, 0, stream>>>(
      reps_ws, wo, bo, out);
}
